// Model_14199161880701
// MI455X (gfx1250) — compile-verified, matrix operand order checked by reading, not run on hardware
//
#include <hip/hip_runtime.h>
#include <hip/hip_bf16.h>
#include <stdint.h>

// Problem constants (from reference)
#define V_   32000
#define H_   512
#define E_   300
#define B_   32
#define T_   32
#define S_   400
#define OOV_ 50
#define VE_  (V_ + OOV_)          // 32050
#define NEG_INF (-1000000000000.0f)

typedef __attribute__((ext_vector_type(16))) __bf16 v16bf;
typedef __attribute__((ext_vector_type(8)))  float  v8f;

// ---- optional CDNA5 async global->LDS path (probe via __has_builtin) ----
#if defined(__HIP_DEVICE_COMPILE__) && defined(__has_builtin)
#if __has_builtin(__builtin_amdgcn_global_load_async_to_lds_b128)
#define HAVE_ASYNC_LDS 1
#endif
#if __has_builtin(__builtin_amdgcn_s_wait_asynccnt)
#define HAVE_WAIT_ASYNC 1
#endif
#endif
#ifndef HAVE_ASYNC_LDS
#define HAVE_ASYNC_LDS 0
#endif
#ifndef HAVE_WAIT_ASYNC
#define HAVE_WAIT_ASYNC 0
#endif

#if HAVE_ASYNC_LDS
typedef int v4i_ __attribute__((vector_size(16)));
typedef __attribute__((address_space(1))) v4i_ as1_v4i;
typedef __attribute__((address_space(3))) v4i_ as3_v4i;
#endif

// ---------- helpers ----------
__device__ __forceinline__ unsigned short f2bf(float x) {
    unsigned int u = __float_as_uint(x);
    unsigned int r = (u + 0x7FFFu + ((u >> 16) & 1u)) >> 16;   // RNE truncate
    return (unsigned short)r;
}

__device__ __forceinline__ void atomicMaxFloat(float* addr, float val) {
    unsigned int* ua = (unsigned int*)addr;
    unsigned int old = *ua;
    while (__uint_as_float(old) < val) {
        unsigned int assumed = old;
        old = atomicCAS(ua, assumed, __float_as_uint(val));
        if (old == assumed) break;
    }
}

__device__ __forceinline__ float sigm(float x) { return 1.0f / (1.0f + expf(-x)); }

// ---------- generic WMMA swizzle kernels ----------
// A layout: flat index = ((mt*KT + kt)*32 + lane)*16 + j
//   element (m = mt*16 + lane%16, k = kt*32 + kmap(j, lane/16)), zero-padded.
// Row m is gathered from concat(s0 row [len0], s1 row [len1]).
__global__ void swzA2_kernel(const float* __restrict__ s0, int str0, int len0,
                             const float* __restrict__ s1, int str1, int len1,
                             unsigned short* __restrict__ dst,
                             int Mtiles, int KT) {
    int idx = blockIdx.x * blockDim.x + threadIdx.x;
    int total = Mtiles * KT * 512;
    if (idx >= total) return;
    int j    = idx & 15;
    int lane = (idx >> 4) & 31;
    int rest = idx >> 9;
    int kt = rest % KT;
    int mt = rest / KT;
    int g = lane >> 4;
    int m = mt * 16 + (lane & 15);
    int k = kt * 32 + ((j < 8 ? j : j + 8) + 8 * g);
    float v = 0.0f;
    if (k < len0)              v = s0[(size_t)m * str0 + k];
    else if (k < len0 + len1)  v = s1[(size_t)m * str1 + (k - len0)];
    dst[idx] = f2bf(v);
}

// B layout: flat index = ((nt*KT + kt)*32 + lane)*16 + j
//   element B[k][n] = W[n][k], n = nt*16 + lane%16; zero-padded for n>=N / k>=K.
__global__ void swzB2_kernel(const float* __restrict__ s0, int str0, int len0,
                             const float* __restrict__ s1, int str1, int len1,
                             unsigned short* __restrict__ dst,
                             int Ntiles, int KT, int N) {
    int idx = blockIdx.x * blockDim.x + threadIdx.x;
    int total = Ntiles * KT * 512;
    if (idx >= total) return;
    int j    = idx & 15;
    int lane = (idx >> 4) & 31;
    int rest = idx >> 9;
    int kt = rest % KT;
    int nt = rest / KT;
    int g = lane >> 4;
    int n = nt * 16 + (lane & 15);
    int k = kt * 32 + ((j < 8 ? j : j + 8) + 8 * g);
    float v = 0.0f;
    if (n < N) {
        if (k < len0)             v = s0[(size_t)n * str0 + k];
        else if (k < len0 + len1) v = s1[(size_t)n * str1 + (k - len0)];
    }
    dst[idx] = f2bf(v);
}

// ---------- generic WMMA GEMM: out[m,n] = A(Mx K) x B(K x N) + bias (+tanh) ----------
__global__ void __launch_bounds__(256) gemm_wmma_generic(
        const unsigned short* __restrict__ Asw,
        const unsigned short* __restrict__ Bsw,
        const float* __restrict__ bias0,
        const float* __restrict__ bias1,
        float* __restrict__ out,
        int KT, int Ntiles, int N, int ldo, int act) {
    int wave = threadIdx.x >> 5;
    int lane = threadIdx.x & 31;
    int nt = blockIdx.x * 8 + wave;
    if (nt >= Ntiles) return;
    int mt = blockIdx.y;

    const v16bf* Ab = (const v16bf*)Asw + (size_t)mt * KT * 32 + lane;
    const v16bf* Bb = (const v16bf*)Bsw + (size_t)nt * KT * 32 + lane;

    v8f acc = {0.f, 0.f, 0.f, 0.f, 0.f, 0.f, 0.f, 0.f};
#pragma unroll 2
    for (int kt = 0; kt < KT; ++kt) {
        v16bf a  = Ab[kt * 32];
        v16bf bb = Bb[kt * 32];
        acc = __builtin_amdgcn_wmma_f32_16x16x32_bf16(
            false, a, false, bb, (short)0, acc, false, false);
    }

    int n = nt * 16 + (lane & 15);
    if (n >= N) return;
    float bv = (bias0 ? bias0[n] : 0.0f) + (bias1 ? bias1[n] : 0.0f);
    int mh = (lane >> 4) * 8;
#pragma unroll
    for (int r = 0; r < 8; ++r) {
        int m = mt * 16 + mh + r;
        float v = acc[r] + bv;
        if (act) v = tanhf(v);
        out[(size_t)m * ldo + n] = v;
    }
}

// ---------- setup / small kernels ----------
__global__ void init_kernel(const float* __restrict__ init_states,
                            float* __restrict__ h0, float* __restrict__ c0,
                            float* __restrict__ ctx, float* __restrict__ cov) {
    int idx = blockIdx.x * blockDim.x + threadIdx.x;
    if (idx < B_ * H_) {
        float v = init_states[idx];
        h0[idx] = v;
        c0[idx] = v;
        ctx[idx] = 0.0f;
    }
    if (idx < B_ * S_) cov[idx] = 0.0f;
}

__global__ void fillneg_kernel(float* __restrict__ ptrbuf) {
    int idx = blockIdx.x * blockDim.x + threadIdx.x;
    if (idx < B_ * VE_) ptrbuf[idx] = NEG_INF;
}

// LSTM pointwise: gates[b, 4*H] -> h_new, c_new
__global__ void lstm_pointwise_kernel(const float* __restrict__ gates,
                                      const float* __restrict__ c_old,
                                      float* __restrict__ h_new,
                                      float* __restrict__ c_new) {
    int idx = blockIdx.x * blockDim.x + threadIdx.x;
    if (idx >= B_ * H_) return;
    int j = idx & (H_ - 1);
    int b = idx >> 9;
    const float* g = gates + (size_t)b * (4 * H_);
    float i_ = sigm(g[j]);
    float f_ = sigm(g[H_ + j]);
    float gg = tanhf(g[2 * H_ + j]);
    float o_ = sigm(g[3 * H_ + j]);
    float cn = f_ * c_old[idx] + i_ * gg;
    c_new[idx] = cn;
    h_new[idx] = o_ * tanhf(cn);
}

// attention: energy -> mask -> scatter-max -> softmax -> ctx; emits attn/cov/energy
__global__ void __launch_bounds__(512) attn_kernel(
        const float* __restrict__ h_new, const float* __restrict__ mem,
        const int* __restrict__ enc_mask, const int* __restrict__ ext_src,
        float* __restrict__ ptrbuf, float* __restrict__ cov, float* __restrict__ ctx,
        float* __restrict__ out_attn, float* __restrict__ out_cov,
        float* __restrict__ out_energy, int t) {
    __shared__ float e_sh[S_];
    __shared__ float a_sh[S_];
    __shared__ float r_sh[512];
    int b = blockIdx.x;
    int tid = threadIdx.x;
    const float* hv = h_new + (size_t)b * H_;

    for (int s = tid; s < S_; s += 512) {
        const float* m = mem + ((size_t)b * S_ + s) * H_;
        float acc = 0.0f;
        for (int k = 0; k < H_; ++k) acc += hv[k] * m[k];
        if (enc_mask[b * S_ + s] == 0) acc = NEG_INF;
        e_sh[s] = acc;
        out_energy[((size_t)t * B_ + b) * S_ + s] = acc;
        atomicMaxFloat(&ptrbuf[(size_t)b * VE_ + ext_src[b * S_ + s]], acc);
    }
    __syncthreads();

    float lm = -3.4e38f;
    for (int s = tid; s < S_; s += 512) lm = fmaxf(lm, e_sh[s]);
    r_sh[tid] = lm; __syncthreads();
    for (int off = 256; off > 0; off >>= 1) {
        if (tid < off) r_sh[tid] = fmaxf(r_sh[tid], r_sh[tid + off]);
        __syncthreads();
    }
    float gmax = r_sh[0]; __syncthreads();

    float ls = 0.0f;
    for (int s = tid; s < S_; s += 512) {
        float ex = expf(e_sh[s] - gmax);
        a_sh[s] = ex;
        ls += ex;
    }
    r_sh[tid] = ls; __syncthreads();
    for (int off = 256; off > 0; off >>= 1) {
        if (tid < off) r_sh[tid] += r_sh[tid + off];
        __syncthreads();
    }
    float gsum = r_sh[0]; __syncthreads();

    for (int s = tid; s < S_; s += 512) {
        float a = a_sh[s] / gsum;
        a_sh[s] = a;
        out_attn[((size_t)t * B_ + b) * S_ + s] = a;
        float cv = cov[b * S_ + s];
        out_cov[((size_t)t * B_ + b) * S_ + s] = cv;   // coverage BEFORE update
        cov[b * S_ + s] = cv + a;
    }
    __syncthreads();

    for (int h = tid; h < H_; h += 512) {
        float acc = 0.0f;
        const float* mb = mem + (size_t)b * S_ * H_ + h;
        for (int s = 0; s < S_; ++s) acc += a_sh[s] * mb[(size_t)s * H_];
        ctx[(size_t)b * H_ + h] = acc;
    }
}

// ---------- vocab GEMM: A(32x512) x Wlog^T(512x32000), dual-M per wave, A in LDS ----------
__global__ void __launch_bounds__(256) logit_wmma_kernel(
        const unsigned short* __restrict__ Asw,   // 16384 ushorts = 32 KB
        const unsigned short* __restrict__ Wsw,
        const float* __restrict__ b_log,
        const float* __restrict__ ptrbuf,
        float* __restrict__ out_logits, int t) {
    __shared__ __align__(32) unsigned short Ash[2 * 16 * 32 * 16];
    int tid = threadIdx.x;

#if HAVE_ASYNC_LDS
    // 256 lanes x 16B per issue, 8 issues = 32 KB
#pragma unroll
    for (int i = 0; i < 8; ++i) {
        __builtin_amdgcn_global_load_async_to_lds_b128(
            (as1_v4i*)(Asw + (size_t)(i * 256 + tid) * 8),
            (as3_v4i*)(Ash + (size_t)(i * 256 + tid) * 8), 0, 0);
    }
#if HAVE_WAIT_ASYNC
    __builtin_amdgcn_s_wait_asynccnt(0);
#else
    asm volatile("s_wait_asynccnt 0x0" ::: "memory");
#endif
#else
    for (int i = tid; i < 2048; i += 256) {
        ((uint4*)Ash)[i] = ((const uint4*)Asw)[i];
    }
#endif
    __syncthreads();

    int wave = tid >> 5;
    int lane = tid & 31;
    int nt = blockIdx.x * 8 + wave;              // 0..1999

    const v16bf* Bb = (const v16bf*)Wsw + (size_t)nt * 16 * 32 + lane;
    const v16bf* A0 = (const v16bf*)Ash + lane;            // mt = 0
    const v16bf* A1 = (const v16bf*)Ash + 16 * 32 + lane;  // mt = 1

    v8f acc0 = {0.f, 0.f, 0.f, 0.f, 0.f, 0.f, 0.f, 0.f};
    v8f acc1 = {0.f, 0.f, 0.f, 0.f, 0.f, 0.f, 0.f, 0.f};
#pragma unroll
    for (int kt = 0; kt < 16; ++kt) {
        v16bf bb = Bb[kt * 32];
        v16bf a0 = A0[kt * 32];
        v16bf a1 = A1[kt * 32];
        acc0 = __builtin_amdgcn_wmma_f32_16x16x32_bf16(
            false, a0, false, bb, (short)0, acc0, false, false);
        acc1 = __builtin_amdgcn_wmma_f32_16x16x32_bf16(
            false, a1, false, bb, (short)0, acc1, false, false);
    }

    int n  = nt * 16 + (lane & 15);
    int mh = (lane >> 4) * 8;
    float bl = b_log[n];
#pragma unroll
    for (int r = 0; r < 8; ++r) {
        // mt = 0
        int b0 = mh + r;
        float pv = ptrbuf[(size_t)b0 * VE_ + n];
        float o  = (pv == NEG_INF) ? 0.0f : pv;
        float le = acc0[r] + bl + o;
        le = (le == 0.0f) ? NEG_INF : le;
        out_logits[((size_t)t * B_ + b0) * VE_ + n] = le;
        // mt = 1
        int b1 = 16 + mh + r;
        pv = ptrbuf[(size_t)b1 * VE_ + n];
        o  = (pv == NEG_INF) ? 0.0f : pv;
        le = acc1[r] + bl + o;
        le = (le == 0.0f) ? NEG_INF : le;
        out_logits[((size_t)t * B_ + b1) * VE_ + n] = le;
    }
}

// last 50 extended-vocab columns (ext part is zero)
__global__ void tail_kernel(const float* __restrict__ ptrbuf,
                            float* __restrict__ out_logits, int t) {
    int idx = blockIdx.x * blockDim.x + threadIdx.x;
    if (idx >= B_ * OOV_) return;
    int i = idx % OOV_;
    int b = idx / OOV_;
    int v = V_ + i;
    float pv = ptrbuf[(size_t)b * VE_ + v];
    float o  = (pv == NEG_INF) ? 0.0f : pv;
    float le = (o == 0.0f) ? NEG_INF : o;
    out_logits[((size_t)t * B_ + b) * VE_ + v] = le;
}

extern "C" void kernel_launch(void* const* d_in, const int* in_sizes, int n_in,
                              void* d_out, int out_size, void* d_ws, size_t ws_size,
                              hipStream_t stream) {
    const float* trg      = (const float*)d_in[0];
    const int*   ext_src  = (const int*)  d_in[1];
    const float* init_st  = (const float*)d_in[2];
    const float* enc      = (const float*)d_in[3];
    const int*   enc_mask = (const int*)  d_in[4];
    const float* W_enc    = (const float*)d_in[5];
    const float* b_enc    = (const float*)d_in[6];
    const float* W_red    = (const float*)d_in[7];
    const float* b_red    = (const float*)d_in[8];
    const float* W_ih     = (const float*)d_in[9];
    const float* W_hh     = (const float*)d_in[10];
    const float* b_ih     = (const float*)d_in[11];
    const float* b_hh     = (const float*)d_in[12];
    const float* W_cat    = (const float*)d_in[13];
    const float* b_cat    = (const float*)d_in[14];
    const float* W_log    = (const float*)d_in[15];
    const float* b_log    = (const float*)d_in[16];

    char* ws = (char*)d_ws;
    size_t off = 0;
    auto alloc = [&](size_t bytes) -> char* {
        char* p = ws + off;
        off = (off + bytes + 255) & ~(size_t)255;
        return p;
    };

    float* mem    = (float*)alloc((size_t)B_ * S_ * H_ * 4);   // 26.2 MB
    float* ptrbuf = (float*)alloc((size_t)B_ * VE_ * 4);       //  4.1 MB
    float* hb[2]  = { (float*)alloc(B_ * H_ * 4), (float*)alloc(B_ * H_ * 4) };
    float* cb[2]  = { (float*)alloc(B_ * H_ * 4), (float*)alloc(B_ * H_ * 4) };
    float* ctx    = (float*)alloc(B_ * H_ * 4);
    float* cov    = (float*)alloc(B_ * S_ * 4);
    float* xbuf   = (float*)alloc(B_ * E_ * 4);
    float* gates  = (float*)alloc(B_ * 4 * H_ * 4);
    float* lin    = (float*)alloc(B_ * H_ * 4);

    unsigned short* Aenc = (unsigned short*)alloc((size_t)800 * 16 * 512 * 2);  // 13.1 MB
    unsigned short* Benc = (unsigned short*)alloc((size_t)32  * 16 * 512 * 2);
    unsigned short* Bred = (unsigned short*)alloc((size_t)19  * 26 * 512 * 2);
    unsigned short* Bihh = (unsigned short*)alloc((size_t)128 * 26 * 512 * 2);  //  3.4 MB
    unsigned short* Bcat = (unsigned short*)alloc((size_t)32  * 32 * 512 * 2);
    unsigned short* Blog = (unsigned short*)alloc((size_t)2000* 16 * 512 * 2);  // 32.8 MB
    unsigned short* Axr  = (unsigned short*)alloc((size_t)2 * 26 * 512 * 2);
    unsigned short* Ag   = (unsigned short*)alloc((size_t)2 * 26 * 512 * 2);
    unsigned short* Ac   = (unsigned short*)alloc((size_t)2 * 32 * 512 * 2);
    unsigned short* Avoc = (unsigned short*)alloc((size_t)2 * 16 * 512 * 2);

    float* out        = (float*)d_out;
    float* out_logits = out;
    float* out_attn   = out_logits + (size_t)T_ * B_ * VE_;
    float* out_cov    = out_attn   + (size_t)T_ * B_ * S_;
    float* out_energy = out_cov    + (size_t)T_ * B_ * S_;

    #define GR(total) (((total) + 255) / 256), 256, 0, stream

    // ---- setup: weight swizzles + memories GEMM ----
    swzA2_kernel<<<GR(800 * 16 * 512)>>>(enc, H_, H_, nullptr, 0, 0, Aenc, 800, 16);
    swzB2_kernel<<<GR(32 * 16 * 512)>>>(W_enc, H_, H_, nullptr, 0, 0, Benc, 32, 16, H_);
    gemm_wmma_generic<<<dim3(4, 800), 256, 0, stream>>>(Aenc, Benc, b_enc, nullptr,
                                                        mem, 16, 32, H_, H_, 0);
    swzB2_kernel<<<GR(19 * 26 * 512)>>>(W_red, E_ + H_, E_ + H_, nullptr, 0, 0,
                                        Bred, 19, 26, E_);
    swzB2_kernel<<<GR(128 * 26 * 512)>>>(W_ih, E_, E_, W_hh, H_, H_,
                                         Bihh, 128, 26, 4 * H_);
    swzB2_kernel<<<GR(32 * 32 * 512)>>>(W_cat, 2 * H_, 2 * H_, nullptr, 0, 0,
                                        Bcat, 32, 32, H_);
    swzB2_kernel<<<GR(2000 * 16 * 512)>>>(W_log, H_, H_, nullptr, 0, 0,
                                          Blog, 2000, 16, V_);
    init_kernel<<<GR(B_ * H_)>>>(init_st, hb[0], cb[0], ctx, cov);

    // ---- sequential decode ----
    for (int t = 0; t < T_; ++t) {
        float* h_old = hb[t & 1];
        float* h_new = hb[(t & 1) ^ 1];
        float* c_old = cb[t & 1];
        float* c_new = cb[(t & 1) ^ 1];

        // x = [emb_t, prev_ctx] @ W_red^T + b_red
        swzA2_kernel<<<GR(2 * 26 * 512)>>>(trg + (size_t)t * E_, T_ * E_, E_,
                                           ctx, H_, H_, Axr, 2, 26);
        gemm_wmma_generic<<<dim3(3, 2), 256, 0, stream>>>(Axr, Bred, b_red, nullptr,
                                                          xbuf, 26, 19, E_, E_, 0);
        // gates = [x, h] @ [W_ih | W_hh]^T + b_ih + b_hh
        swzA2_kernel<<<GR(2 * 26 * 512)>>>(xbuf, E_, E_, h_old, H_, H_, Ag, 2, 26);
        gemm_wmma_generic<<<dim3(16, 2), 256, 0, stream>>>(Ag, Bihh, b_ih, b_hh,
                                                           gates, 26, 128, 4 * H_,
                                                           4 * H_, 0);
        lstm_pointwise_kernel<<<GR(B_ * H_)>>>(gates, c_old, h_new, c_new);

        // pointer scatter buffer + attention
        fillneg_kernel<<<GR(B_ * VE_)>>>(ptrbuf);
        attn_kernel<<<B_, 512, 0, stream>>>(h_new, mem, enc_mask, ext_src,
                                            ptrbuf, cov, ctx,
                                            out_attn, out_cov, out_energy, t);

        // logit_in = tanh([h_new, ctx] @ W_cat^T + b_cat)
        swzA2_kernel<<<GR(2 * 32 * 512)>>>(h_new, H_, H_, ctx, H_, H_, Ac, 2, 32);
        gemm_wmma_generic<<<dim3(4, 2), 256, 0, stream>>>(Ac, Bcat, b_cat, nullptr,
                                                          lin, 32, 32, H_, H_, 1);

        // vocab projection + pointer epilogue
        swzA2_kernel<<<GR(2 * 16 * 512)>>>(lin, H_, H_, nullptr, 0, 0, Avoc, 2, 16);
        logit_wmma_kernel<<<250, 256, 0, stream>>>(Avoc, Blog, b_log, ptrbuf,
                                                   out_logits, t);
        tail_kernel<<<GR(B_ * OOV_)>>>(ptrbuf, out_logits, t);
    }
    #undef GR
}
